// Compiler_58420145160287
// MI455X (gfx1250) — compile-verified, matrix operand order checked by reading, not run on hardware
//
#include <hip/hip_runtime.h>
#include <hip/hip_bf16.h>
#include <math.h>

typedef __attribute__((ext_vector_type(16))) _Float16 v16h;
typedef __attribute__((ext_vector_type(8)))  _Float16 v8h;
typedef __attribute__((ext_vector_type(8)))  float    v8f;

#define B_   8
#define T_   512
#define H_   1024
#define DIN_ 1032
#define KP_  1056   // DIN padded to multiple of 32 for WMMA K-chunks
#define V_   8
#define L_   4
#define VH_  (V_*H_)   // 8192 = combined (v,h) contraction dim for the fused mix GEMM

#define LDS_STRIDE 40   // halfs per row: 80B, 16B-aligned, bank-spread

// branch-free exact-GELU: erf via Abramowitz-Stegun 7.1.26 (|err| < 1.5e-7)
__device__ __forceinline__ float gelu_exact(float x) {
  float u  = 0.70710678118654752f * x;
  float au = fabsf(u);
  float t  = 1.0f / fmaf(0.3275911f, au, 1.0f);
  float p  = t * fmaf(t, fmaf(t, fmaf(t, fmaf(t, 1.061405429f, -1.453152027f),
                                      1.421413741f), -0.284496736f), 0.254829592f);
  float er = fmaf(-p, __expf(-au*au), 1.0f);
  er = (u < 0.0f) ? -er : er;
  return 0.5f * x * (1.0f + er);
}

// ---------------- small scalar kernels ----------------

// per (b,t) row: tt(6), pol(2) heads; also emit fp16 x_aug row (padded to KP_)
__global__ __launch_bounds__(256) void k_concept_xaug(
    const float* __restrict__ x, const float* __restrict__ Wtt, const float* __restrict__ btt,
    const float* __restrict__ Wpol, const float* __restrict__ bpol,
    float* __restrict__ tt_out, float* __restrict__ pol_out, _Float16* __restrict__ xaug)
{
  __shared__ float xrow[H_];
  int n = blockIdx.x;
  const float* xr = x + (size_t)n * H_;
  for (int it = 0; it < 4; ++it) xrow[threadIdx.x + it*256] = xr[threadIdx.x + it*256];
  __syncthreads();
  int wave = threadIdx.x >> 5, lane = threadIdx.x & 31;
  const float* wrow = (wave < 6) ? (Wtt + (size_t)wave * H_) : (Wpol + (size_t)(wave-6) * H_);
  float p = 0.f;
  for (int d = lane; d < H_; d += 32) p += xrow[d] * wrow[d];
  for (int m = 16; m >= 1; m >>= 1) p += __shfl_xor(p, m, 32);
  _Float16* xa = xaug + (size_t)n * KP_;
  if (lane == 0) {
    float val = p + ((wave < 6) ? btt[wave] : bpol[wave-6]);
    if (wave < 6) tt_out[(size_t)n*6 + wave] = val;
    else          pol_out[(size_t)n*2 + (wave-6)] = val;
    xa[H_ + wave] = (_Float16)val;
  }
  for (int it = 0; it < 4; ++it) { int d = threadIdx.x + it*256; xa[d] = (_Float16)xrow[d]; }
  if (threadIdx.x < (KP_ - DIN_)) xa[DIN_ + threadIdx.x] = (_Float16)0.f;
}

__global__ __launch_bounds__(256) void k_conv_w1(const float* __restrict__ W1,
                                                 _Float16* __restrict__ W1h, int total)
{
  int i = blockIdx.x*256 + threadIdx.x;
  if (i >= total) return;
  int d  = i % KP_;
  int vh = i / KP_;
  W1h[i] = (d < DIN_) ? (_Float16)W1[(size_t)vh*DIN_ + d] : (_Float16)0.f;
}

// W2 (V,H,H)[v][o][h] -> W2all fp16 [o][v*H+h]  (rows of the fused K=8192 GEMM B-matrix)
__global__ __launch_bounds__(256) void k_conv_w2t(const float* __restrict__ W2,
                                                  _Float16* __restrict__ W2all, int total)
{
  int i = blockIdx.x*256 + threadIdx.x;
  if (i >= total) return;
  int h  = i & (H_-1);
  int o  = (i >> 10) & (H_-1);
  int v  = i >> 20;
  W2all[(size_t)o*VH_ + v*H_ + h] = (_Float16)W2[i];
}

// pooled[b,h] = mean_t x ; also fill context[:,0:H]
__global__ __launch_bounds__(256) void k_pool(const float* __restrict__ x,
                                              float* __restrict__ pooled_out,
                                              float* __restrict__ context)
{
  int idx = blockIdx.x*256 + threadIdx.x;     // B*H
  int b = idx >> 10, h = idx & 1023;
  const float* xp = x + (size_t)b*T_*H_ + h;
  float s = 0.f;
  for (int t = 0; t < T_; ++t) s += xp[(size_t)t*H_];
  float m = s * (1.0f/T_);
  pooled_out[idx] = m;
  context[(size_t)b*DIN_ + h] = m;
}

// concept_summary means -> context tail
__global__ void k_summary(const float* __restrict__ tt, const float* __restrict__ pol,
                          float* __restrict__ context)
{
  int t = threadIdx.x; if (t >= B_*8) return;
  int b = t >> 3, j = t & 7;
  float s = 0.f;
  if (j < 6) { const float* p = tt  + (size_t)b*T_*6 + j;     for (int i=0;i<T_;++i) s += p[(size_t)i*6]; }
  else       { const float* p = pol + (size_t)b*T_*2 + (j-6); for (int i=0;i<T_;++i) s += p[(size_t)i*2]; }
  context[(size_t)b*DIN_ + H_ + j] = s * (1.0f/T_);
}

__global__ __launch_bounds__(256) void k_h0(const float* __restrict__ context,
                                            const float* __restrict__ Wc2h,
                                            const float* __restrict__ bc2h, float* __restrict__ h)
{
  int idx = blockIdx.x*256 + threadIdx.x;     // B*H
  int b = idx >> 10, i = idx & 1023;
  const float* c = context + (size_t)b*DIN_;
  const float* w = Wc2h + (size_t)i*DIN_;
  float a = bc2h[i];
  for (int d = 0; d < DIN_; ++d) a += c[d]*w[d];
  h[idx] = a;
}

// one GRU step (zero input -> gi = b_ih)
__global__ __launch_bounds__(256) void k_gru_step(const float* __restrict__ hin,
    const float* __restrict__ Whh, const float* __restrict__ bhh, const float* __restrict__ bih,
    float* __restrict__ hout, float* __restrict__ hs_l)
{
  int idx = blockIdx.x*256 + threadIdx.x;     // B*H
  int b = idx >> 10, i = idx & 1023;
  const float* hr = hin + (size_t)b*H_;
  const float* wr = Whh + (size_t)i*H_;
  const float* wz = Whh + (size_t)(H_+i)*H_;
  const float* wn = Whh + (size_t)(2*H_+i)*H_;
  float gr = bhh[i], gz = bhh[H_+i], gn = bhh[2*H_+i];
  for (int d = 0; d < H_; ++d) { float hd = hr[d]; gr += hd*wr[d]; gz += hd*wz[d]; gn += hd*wn[d]; }
  float r  = 1.f/(1.f+expf(-(bih[i]      + gr)));
  float z  = 1.f/(1.f+expf(-(bih[H_+i]   + gz)));
  float nn = tanhf(bih[2*H_+i] + r*gn);
  float hn = (1.f - z)*nn + z*hr[i];
  hout[idx] = hn; hs_l[idx] = hn;
}

// program_logits[b,l,v]
__global__ void k_logits(const float* __restrict__ hs, const float* __restrict__ Wproj,
                         const float* __restrict__ bproj, float* __restrict__ logits)
{
  int idx = threadIdx.x;                      // B*L*V = 256
  int b = idx >> 5, l = (idx >> 3) & 3, v = idx & 7;
  const float* h = hs + ((size_t)l*B_ + b)*H_;
  const float* w = Wproj + (size_t)v*H_;
  float a = bproj[v];
  for (int d = 0; d < H_; ++d) a += h[d]*w[d];
  logits[idx] = a;
}

// w[b,v] = mean_l softmax(logits[b,l,:])[v]
__global__ void k_softmax_w(const float* __restrict__ logits, float* __restrict__ wbuf)
{
  int t = threadIdx.x; if (t >= B_*V_) return;
  int b = t >> 3, v = t & 7;
  float acc = 0.f;
  for (int l = 0; l < L_; ++l) {
    const float* row = logits + ((size_t)b*L_ + l)*V_;
    float m = row[0];
    for (int k = 1; k < V_; ++k) m = fmaxf(m, row[k]);
    float s = 0.f;
    for (int k = 0; k < V_; ++k) s += expf(row[k]-m);
    acc += expf(row[v]-m)/s;
  }
  wbuf[t] = acc * (1.0f/L_);
}

// biasmix[b,o] = sum_v w[b,v]*b2[v,o]
__global__ __launch_bounds__(256) void k_biasmix(const float* __restrict__ wbuf,
                                                 const float* __restrict__ b2,
                                                 float* __restrict__ biasmix)
{
  int idx = blockIdx.x*256 + threadIdx.x;   // B*H
  int b = idx >> 10, o = idx & 1023;
  float s = 0.f;
  for (int v = 0; v < V_; ++v) s += wbuf[b*V_+v] * b2[(size_t)v*H_ + o];
  biasmix[idx] = s;
}

// ---------------- WMMA GEMM 1: hmidAll[n, v*H+o] = w[b,v]*GELU(xaug·W1[v]^T + b1[v]) ----------------
__global__ __launch_bounds__(256, 1) void k_gemm1_wmma(
    const _Float16* __restrict__ A, const _Float16* __restrict__ W1h,
    const float* __restrict__ b1, const float* __restrict__ wbuf,
    _Float16* __restrict__ HmidAll)
{
  __shared__ _Float16 ldsA[2][128*LDS_STRIDE];
  __shared__ _Float16 ldsB[2][128*LDS_STRIDE];
  const int K = KP_;
  const int vop = blockIdx.z;
  const _Float16* Bw = W1h + (size_t)vop * H_ * KP_;
  const int tid  = threadIdx.x;
  const int bm   = blockIdx.y * 128, bn = blockIdx.x * 128;
  const int wave = tid >> 5, lane = tid & 31;
  const int wm   = (wave >> 2) * 64;
  const int wn   = (wave & 3) * 32;
  v8f acc[4][2] = {};
  const int lrow = lane & 15;
  const int kb0  = (lane < 16) ? 0 : 8;

  const int r0 = (tid      ) >> 2, s0 = (tid      ) & 3;
  const int r1 = (tid + 256) >> 2, s1 = (tid + 256) & 3;
  const _Float16* gA0 = A  + (size_t)(bm + r0)*K + s0*8;
  const _Float16* gA1 = A  + (size_t)(bm + r1)*K + s1*8;
  const _Float16* gB0 = Bw + (size_t)(bn + r0)*K + s0*8;
  const _Float16* gB1 = Bw + (size_t)(bn + r1)*K + s1*8;
  const int dA0 = r0*LDS_STRIDE + s0*8, dA1 = r1*LDS_STRIDE + s1*8;
  const int nk = K >> 5;

  uint4 ra0 = *(const uint4*)(gA0), ra1 = *(const uint4*)(gA1);
  uint4 rb0 = *(const uint4*)(gB0), rb1 = *(const uint4*)(gB1);
  *(uint4*)(&ldsA[0][dA0]) = ra0;  *(uint4*)(&ldsA[0][dA1]) = ra1;
  *(uint4*)(&ldsB[0][dA0]) = rb0;  *(uint4*)(&ldsB[0][dA1]) = rb1;
  __syncthreads();

  for (int kc = 0; kc < nk; ++kc) {
    const int buf = kc & 1;
    const bool more = (kc + 1) < nk;
    if (more) {
      const int kk = (kc + 1) << 5;
      ra0 = *(const uint4*)(gA0 + kk); ra1 = *(const uint4*)(gA1 + kk);
      rb0 = *(const uint4*)(gB0 + kk); rb1 = *(const uint4*)(gB1 + kk);
    }
    v16h af[4], bf[2];
#pragma unroll
    for (int mf = 0; mf < 4; ++mf) {
      const _Float16* p = &ldsA[buf][(wm + mf*16 + lrow)*LDS_STRIDE + kb0];
      union { v16h v; v8h h[2]; } u;
      u.h[0] = *(const v8h*)(p); u.h[1] = *(const v8h*)(p + 16);
      af[mf] = u.v;
    }
#pragma unroll
    for (int nf = 0; nf < 2; ++nf) {
      const _Float16* p = &ldsB[buf][(wn + nf*16 + lrow)*LDS_STRIDE + kb0];
      union { v16h v; v8h h[2]; } u;
      u.h[0] = *(const v8h*)(p); u.h[1] = *(const v8h*)(p + 16);
      bf[nf] = u.v;
    }
#pragma unroll
    for (int mf = 0; mf < 4; ++mf)
#pragma unroll
      for (int nf = 0; nf < 2; ++nf)
        acc[mf][nf] = __builtin_amdgcn_wmma_f32_16x16x32_f16(
            false, af[mf], false, bf[nf], (short)0, acc[mf][nf], false, false);
    if (more) {
      *(uint4*)(&ldsA[1-buf][dA0]) = ra0;  *(uint4*)(&ldsA[1-buf][dA1]) = ra1;
      *(uint4*)(&ldsB[1-buf][dA0]) = rb0;  *(uint4*)(&ldsB[1-buf][dA1]) = rb1;
    }
    __syncthreads();
  }

  const int lcol  = lane & 15;
  const int rbase = (lane < 16) ? 0 : 8;
  const float wv  = wbuf[(bm >> 9) * V_ + vop];   // 128-row tile stays in one batch
#pragma unroll
  for (int mf = 0; mf < 4; ++mf)
#pragma unroll
    for (int nf = 0; nf < 2; ++nf) {
      int col = bn + wn + nf*16 + lcol;
      float bb = b1[vop*H_ + col];
#pragma unroll
      for (int r = 0; r < 8; ++r) {
        int row = bm + wm + mf*16 + rbase + r;
        float g = gelu_exact(acc[mf][nf][r] + bb);
        HmidAll[(size_t)row*VH_ + vop*H_ + col] = (_Float16)(wv * g);
      }
    }
}

// ---------------- WMMA GEMM 2 (fused mix): out = hmidAll(4096x8192)·W2all(1024x8192)^T + biasmix ----------------
__global__ __launch_bounds__(256, 1) void k_gemm2_wmma(
    const _Float16* __restrict__ A, const _Float16* __restrict__ Bw,
    const float* __restrict__ biasmix, float* __restrict__ out)
{
  __shared__ _Float16 ldsA[2][128*LDS_STRIDE];
  __shared__ _Float16 ldsB[2][128*LDS_STRIDE];
  const int K = VH_;     // 8192
  const int tid  = threadIdx.x;
  const int bm   = blockIdx.y * 128, bn = blockIdx.x * 128;
  const int wave = tid >> 5, lane = tid & 31;
  const int wm   = (wave >> 2) * 64;
  const int wn   = (wave & 3) * 32;
  v8f acc[4][2] = {};
  const int lrow = lane & 15;
  const int kb0  = (lane < 16) ? 0 : 8;

  const int r0 = (tid      ) >> 2, s0 = (tid      ) & 3;
  const int r1 = (tid + 256) >> 2, s1 = (tid + 256) & 3;
  const _Float16* gA0 = A  + (size_t)(bm + r0)*K + s0*8;
  const _Float16* gA1 = A  + (size_t)(bm + r1)*K + s1*8;
  const _Float16* gB0 = Bw + (size_t)(bn + r0)*K + s0*8;
  const _Float16* gB1 = Bw + (size_t)(bn + r1)*K + s1*8;
  const int dA0 = r0*LDS_STRIDE + s0*8, dA1 = r1*LDS_STRIDE + s1*8;
  const int nk = K >> 5;  // 256

  uint4 ra0 = *(const uint4*)(gA0), ra1 = *(const uint4*)(gA1);
  uint4 rb0 = *(const uint4*)(gB0), rb1 = *(const uint4*)(gB1);
  *(uint4*)(&ldsA[0][dA0]) = ra0;  *(uint4*)(&ldsA[0][dA1]) = ra1;
  *(uint4*)(&ldsB[0][dA0]) = rb0;  *(uint4*)(&ldsB[0][dA1]) = rb1;
  __syncthreads();

  for (int kc = 0; kc < nk; ++kc) {
    const int buf = kc & 1;
    const bool more = (kc + 1) < nk;
    if (more) {
      const int kk = (kc + 1) << 5;
      ra0 = *(const uint4*)(gA0 + kk); ra1 = *(const uint4*)(gA1 + kk);
      rb0 = *(const uint4*)(gB0 + kk); rb1 = *(const uint4*)(gB1 + kk);
    }
    v16h af[4], bf[2];
#pragma unroll
    for (int mf = 0; mf < 4; ++mf) {
      const _Float16* p = &ldsA[buf][(wm + mf*16 + lrow)*LDS_STRIDE + kb0];
      union { v16h v; v8h h[2]; } u;
      u.h[0] = *(const v8h*)(p); u.h[1] = *(const v8h*)(p + 16);
      af[mf] = u.v;
    }
#pragma unroll
    for (int nf = 0; nf < 2; ++nf) {
      const _Float16* p = &ldsB[buf][(wn + nf*16 + lrow)*LDS_STRIDE + kb0];
      union { v16h v; v8h h[2]; } u;
      u.h[0] = *(const v8h*)(p); u.h[1] = *(const v8h*)(p + 16);
      bf[nf] = u.v;
    }
#pragma unroll
    for (int mf = 0; mf < 4; ++mf)
#pragma unroll
      for (int nf = 0; nf < 2; ++nf)
        acc[mf][nf] = __builtin_amdgcn_wmma_f32_16x16x32_f16(
            false, af[mf], false, bf[nf], (short)0, acc[mf][nf], false, false);
    if (more) {
      *(uint4*)(&ldsA[1-buf][dA0]) = ra0;  *(uint4*)(&ldsA[1-buf][dA1]) = ra1;
      *(uint4*)(&ldsB[1-buf][dA0]) = rb0;  *(uint4*)(&ldsB[1-buf][dA1]) = rb1;
    }
    __syncthreads();
  }

  const int lcol  = lane & 15;
  const int rbase = (lane < 16) ? 0 : 8;
  const float* bmix = biasmix + (size_t)(bm >> 9) * H_;
#pragma unroll
  for (int mf = 0; mf < 4; ++mf)
#pragma unroll
    for (int nf = 0; nf < 2; ++nf) {
      int col = bn + wn + nf*16 + lcol;
      float bb = bmix[col];
#pragma unroll
      for (int r = 0; r < 8; ++r) {
        int row = bm + wm + mf*16 + rbase + r;
        out[(size_t)row*H_ + col] = acc[mf][nf][r] + bb;
      }
    }
}

// ---------------- host launch ----------------

extern "C" void kernel_launch(void* const* d_in, const int* in_sizes, int n_in,
                              void* d_out, int out_size, void* d_ws, size_t ws_size,
                              hipStream_t stream)
{
  (void)in_sizes; (void)n_in; (void)out_size; (void)ws_size;
  const float* x     = (const float*)d_in[0];
  const float* Wtt   = (const float*)d_in[1];
  const float* btt   = (const float*)d_in[2];
  const float* Wpol  = (const float*)d_in[3];
  const float* bpol  = (const float*)d_in[4];
  const float* Wc2h  = (const float*)d_in[5];
  const float* bc2h  = (const float*)d_in[6];
  /* d_in[7] = W_ih: unused (GRU input is zero, gi == b_ih) */
  const float* Whh   = (const float*)d_in[8];
  const float* bih   = (const float*)d_in[9];
  const float* bhh   = (const float*)d_in[10];
  const float* Wproj = (const float*)d_in[11];
  const float* bproj = (const float*)d_in[12];
  const float* W1    = (const float*)d_in[13];
  const float* b1    = (const float*)d_in[14];
  const float* W2    = (const float*)d_in[15];
  const float* b2    = (const float*)d_in[16];

  float* out        = (float*)d_out;                        // (B,T,H)
  float* logits_out = out + (size_t)B_*T_*H_;               // (B,L,V)
  float* tt_out     = logits_out + B_*L_*V_;                // (B,T,6)
  float* pol_out    = tt_out + (size_t)B_*T_*6;             // (B,T,2)
  float* pooled_out = pol_out + (size_t)B_*T_*2;            // (B,H)

  char* wsp = (char*)d_ws;
  size_t off = 0;
  auto carve = [&](size_t bytes) -> void* {
    off = (off + 255) & ~(size_t)255;
    void* p = wsp + off; off += bytes; return p;
  };
  _Float16* xaugh   = (_Float16*)carve((size_t)B_*T_*KP_*2);   // fp16 x_aug, K-padded
  _Float16* W1h     = (_Float16*)carve((size_t)V_*H_*KP_*2);   // fp16 W1, K-padded
  _Float16* W2all   = (_Float16*)carve((size_t)H_*VH_*2);      // fp16 W2, [o][v*H+h]
  _Float16* hmidAll = (_Float16*)carve((size_t)B_*T_*VH_*2);   // fp16 w-scaled GELU mid, all ops
  float* context = (float*)carve((size_t)B_*DIN_*4);
  float* hA      = (float*)carve((size_t)B_*H_*4);
  float* hB      = (float*)carve((size_t)B_*H_*4);
  float* hs      = (float*)carve((size_t)L_*B_*H_*4);
  float* wbuf    = (float*)carve((size_t)B_*V_*4);
  float* biasmix = (float*)carve((size_t)B_*H_*4);

  k_concept_xaug<<<B_*T_, 256, 0, stream>>>(x, Wtt, btt, Wpol, bpol, tt_out, pol_out, xaugh);
  k_conv_w1<<<(V_*H_*KP_ + 255)/256, 256, 0, stream>>>(W1, W1h, V_*H_*KP_);
  k_conv_w2t<<<(V_*H_*H_ + 255)/256, 256, 0, stream>>>(W2, W2all, V_*H_*H_);
  k_pool<<<(B_*H_)/256, 256, 0, stream>>>(x, pooled_out, context);
  k_summary<<<1, 64, 0, stream>>>(tt_out, pol_out, context);
  k_h0<<<(B_*H_)/256, 256, 0, stream>>>(context, Wc2h, bc2h, hA);

  float* hin = hA; float* hout = hB;
  for (int l = 0; l < L_; ++l) {
    k_gru_step<<<(B_*H_)/256, 256, 0, stream>>>(hin, Whh, bhh, bih, hout, hs + (size_t)l*B_*H_);
    float* t = hin; hin = hout; hout = t;
  }
  k_logits<<<1, 256, 0, stream>>>(hs, Wproj, bproj, logits_out);
  k_softmax_w<<<1, 64, 0, stream>>>(logits_out, wbuf);
  k_biasmix<<<(B_*H_)/256, 256, 0, stream>>>(wbuf, b2, biasmix);

  dim3 g1(H_/128, (B_*T_)/128, V_);   // (8, 32, 8)
  k_gemm1_wmma<<<g1, 256, 0, stream>>>(xaugh, W1h, b1, wbuf, hmidAll);
  dim3 g2(H_/128, (B_*T_)/128);       // (8, 32)
  k_gemm2_wmma<<<g2, 256, 0, stream>>>(hmidAll, W2all, biasmix, out);
}
